// RNN_Delay_57140244906265
// MI455X (gfx1250) — matrix operand order repaired and checked by reading, not compiled, not run on hardware
//
#include <hip/hip_runtime.h>
#include <hip/hip_bf16.h>

// RNN: B=64, T=1000, INP=64, HID=512, ACT=2
// d_out layout (floats): out[64*1000*2] | hn_last[64*512] | rnn_out[64*1000*512]

typedef _Float16 v16h __attribute__((ext_vector_type(16)));
typedef _Float16 v8h  __attribute__((ext_vector_type(8)));
typedef float    v8f  __attribute__((ext_vector_type(8)));

#define HID 512
#define BATCH 64
#define TSTEPS 1000
#define LDH 520   // padded f16 row stride for LDS h buffer (bank-conflict-free A loads)

__device__ __forceinline__ v16h cat8(v8h lo, v8h hi) {
    return __builtin_shufflevector(lo, hi, 0,1,2,3,4,5,6,7,8,9,10,11,12,13,14,15);
}

__device__ __forceinline__ v8f wmma_f16(v16h a, v16h b, v8f c) {
    return __builtin_amdgcn_wmma_f32_16x16x32_f16(false, a, false, b, (short)0, c, false, false);
}

// ---------------------------------------------------------------------------
// Pack W_hh (512x512 f32 row-major, B = W[k][n]) into per-lane WMMA B-fragment
// order, f16. Fragment (n-tile n, k-tile kt), lane l holds 16 contiguous f16:
//   b[i] = W[32*kt + 8*(l>>4) + (i&7) + 16*(i>>3)][n*16 + (l&15)]
// dst index = ((n*16 + kt)*32 + l)*16 + i   (512 KB total)
// ---------------------------------------------------------------------------
__global__ void pack_whh_kernel(const float* __restrict__ W, _Float16* __restrict__ dst) {
    int tid = blockIdx.x * blockDim.x + threadIdx.x;       // 0 .. 262143
    int i  = tid & 15;
    int l  = (tid >> 4) & 31;
    int kt = (tid >> 9) & 15;
    int n  = tid >> 13;
    int row = 32 * kt + 8 * (l >> 4) + (i & 7) + 16 * (i >> 3);
    int col = n * 16 + (l & 15);
    dst[tid] = (_Float16)W[row * HID + col];
}

// Same packing for W_ih (64x512): kt in {0,1}. dst = 64 KB.
__global__ void pack_wih_kernel(const float* __restrict__ W, _Float16* __restrict__ dst) {
    int tid = blockIdx.x * blockDim.x + threadIdx.x;       // 0 .. 32767
    int i  = tid & 15;
    int l  = (tid >> 4) & 31;
    int kt = (tid >> 9) & 1;
    int n  = tid >> 10;
    int row = 32 * kt + 8 * (l >> 4) + (i & 7) + 16 * (i >> 3); // < 64
    int col = n * 16 + (l & 15);
    dst[tid] = (_Float16)W[row * HID + col];
}

// ---------------------------------------------------------------------------
// Phase 1: inp_proj = inp @ W_ih  ([64000,64] x [64,512] -> rnn region, f32)
// 8 waves/block, one 16x16 tile per wave, K = 2 x 32.
// ---------------------------------------------------------------------------
__global__ void __launch_bounds__(256) gemm_ih_kernel(const float* __restrict__ inp,
                                                      const _Float16* __restrict__ wihp,
                                                      float* __restrict__ rnn) {
    int wv = threadIdx.x >> 5, lane = threadIdx.x & 31;
    int tile = blockIdx.x * 8 + wv;                        // 0 .. 127999
    int mt = tile >> 5, nt = tile & 31;
    int g = lane >> 4, ln = lane & 15;

    v8f acc = {};
    #pragma unroll
    for (int kt = 0; kt < 2; ++kt) {
        // A fragment: lane holds K-chunks [8g,8g+8) and [16+8g,16+8g+8) of row mt*16+ln
        const float* ap = inp + (mt * 16 + ln) * 64 + kt * 32;
        v16h a;
        #pragma unroll
        for (int i = 0; i < 8; ++i)  a[i] = (_Float16)ap[8 * g + i];
        #pragma unroll
        for (int i = 8; i < 16; ++i) a[i] = (_Float16)ap[8 + 8 * g + i];
        v16h b = *(const v16h*)(wihp + ((nt * 2 + kt) * 32 + lane) * 16);
        acc = wmma_f16(a, b, acc);
    }
    int col = nt * 16 + ln;
    #pragma unroll
    for (int r = 0; r < 8; ++r) {
        int row = mt * 16 + r + 8 * g;                     // C layout: M = r + 8*(lane>=16)
        rnn[row * HID + col] = acc[r];
    }
}

// ---------------------------------------------------------------------------
// Phase 2: persistent single-workgroup scan. 32 waves on one WGP.
// wave w: M-tile m = w>>3 (16 batch rows), N-tiles n = (w&7)*4 + {0..3}.
// h double-buffered in LDS (f16, padded stride); W_hh streamed from L2 (packed).
// x_t read from rnn region and overwritten in place with h_t.
// ---------------------------------------------------------------------------
__global__ void __launch_bounds__(1024, 1) rnn_scan_kernel(const float* __restrict__ hn_in,
                                                           const _Float16* __restrict__ whhp,
                                                           float* __restrict__ rnn,
                                                           float* __restrict__ hn_out) {
    extern __shared__ __align__(16) _Float16 hbuf[];       // [2][64*LDH]
    const int tid = threadIdx.x;
    const int wv = tid >> 5, lane = tid & 31;
    const int g = lane >> 4, ln = lane & 15;
    const int m = wv >> 3;                                 // M-tile 0..3
    const int ng = wv & 7;                                 // N-group 0..7

    // init h0 = hn[0]
    for (int idx = tid; idx < BATCH * HID; idx += 1024) {
        int b = idx >> 9, h = idx & (HID - 1);
        hbuf[b * LDH + h] = (_Float16)hn_in[idx];
    }
    __syncthreads();

    for (int t = 0; t < TSTEPS; ++t) {
        const _Float16* hc = hbuf + (t & 1) * (BATCH * LDH);
        _Float16* hx = hbuf + ((t + 1) & 1) * (BATCH * LDH);

        v8f acc[4];
        #pragma unroll
        for (int j = 0; j < 4; ++j) acc[j] = (v8f){};

        #pragma unroll
        for (int kt = 0; kt < 16; ++kt) {
            const _Float16* ap = hc + (m * 16 + ln) * LDH + kt * 32;
            v8h lo = *(const v8h*)(ap + 8 * g);            // ds_load_b128, conflict-free
            v8h hi = *(const v8h*)(ap + 16 + 8 * g);
            v16h a = cat8(lo, hi);
            #pragma unroll
            for (int j = 0; j < 4; ++j) {
                int n = ng * 4 + j;
                v16h b = *(const v16h*)(whhp + ((n * 16 + kt) * 32 + lane) * 16);
                acc[j] = wmma_f16(a, b, acc[j]);
            }
        }

        #pragma unroll
        for (int j = 0; j < 4; ++j) {
            int col = (ng * 4 + j) * 16 + ln;
            #pragma unroll
            for (int r = 0; r < 8; ++r) {
                int brow = m * 16 + r + 8 * g;
                int idx = (brow * TSTEPS + t) * HID + col;
                float v = acc[j][r] + rnn[idx];            // + x_t (inp_proj)
                float hval = 1.0f / (1.0f + __expf(-v));
                rnn[idx] = hval;                           // rnn_out[b][t][h]
                hx[brow * LDH + col] = (_Float16)hval;
                if (t == TSTEPS - 1) hn_out[brow * HID + col] = hval;
            }
        }
        __syncthreads();
    }
}

// ---------------------------------------------------------------------------
// Phase 3: out = sigmoid(rnn_out @ W_fc + b_fc), ACT=2. One thread per (b,t).
// ---------------------------------------------------------------------------
__global__ void __launch_bounds__(256) fc_out_kernel(const float* __restrict__ rnn,
                                                     const float* __restrict__ Wfc,
                                                     const float* __restrict__ bfc,
                                                     float* __restrict__ out) {
    int bt = blockIdx.x * blockDim.x + threadIdx.x;        // 0 .. 63999
    const float* row = rnn + bt * HID;
    const float2* w2 = (const float2*)Wfc;
    float a0 = 0.f, a1 = 0.f;
    #pragma unroll 4
    for (int k = 0; k < HID; ++k) {
        float h = row[k];
        float2 w = w2[k];
        a0 += h * w.x;
        a1 += h * w.y;
    }
    a0 += bfc[0]; a1 += bfc[1];
    out[bt * 2 + 0] = 1.0f / (1.0f + __expf(-a0));
    out[bt * 2 + 1] = 1.0f / (1.0f + __expf(-a1));
}

extern "C" void kernel_launch(void* const* d_in, const int* in_sizes, int n_in,
                              void* d_out, int out_size, void* d_ws, size_t ws_size,
                              hipStream_t stream) {
    const float* inp = (const float*)d_in[0];   // [64,1000,64]
    const float* hn  = (const float*)d_in[1];   // [1,64,512]
    const float* Whh = (const float*)d_in[2];   // [512,512]
    const float* Wih = (const float*)d_in[3];   // [64,512]
    const float* Wfc = (const float*)d_in[4];   // [512,2]
    const float* bfc = (const float*)d_in[5];   // [2]

    float* out = (float*)d_out;                 // [64,1000,2]
    float* hnl = out + BATCH * TSTEPS * 2;      // [1,64,512]
    float* rnn = hnl + BATCH * HID;             // [64,1000,512]

    _Float16* whhp = (_Float16*)d_ws;           // 512 KB packed W_hh fragments
    _Float16* wihp = whhp + 32 * 16 * 32 * 16;  // 64 KB packed W_ih fragments

    pack_whh_kernel<<<1024, 256, 0, stream>>>(Whh, whhp);
    pack_wih_kernel<<<128, 256, 0, stream>>>(Wih, wihp);
    gemm_ih_kernel<<<16000, 256, 0, stream>>>(inp, wihp, rnn);
    rnn_scan_kernel<<<1, 1024, 2 * BATCH * LDH * sizeof(_Float16), stream>>>(hn, whhp, rnn, hnl);
    fc_out_kernel<<<250, 256, 0, stream>>>(rnn, Wfc, bfc, out);
}
